// DotProductAttention_19138374270990
// MI455X (gfx1250) — matrix operand order repaired and checked by reading, not compiled, not run on hardware
//
#include <hip/hip_runtime.h>
#include <math.h>

// ---------------------------------------------------------------------------
// Flash-attention for MI455X (gfx1250, wave32, WMMA), round 4.
// B=32, S=2048, D=128, fp32 in/out, bf16 WMMA with f32 accumulation.
//
// Fast path (needs 33.5 MB workspace):
//   prepass: K -> bf16 row-major, V -> bf16 transposed [b][d][s] in d_ws.
//   main:    K/V tiles streamed into double-buffered LDS with
//            global_load_async_to_lds_b128 (ASYNCcnt), zero staging VALU.
// Fallback path (small ws): register-pipelined staging.
// ---------------------------------------------------------------------------

typedef __attribute__((ext_vector_type(16))) __bf16 v16bf;
typedef __attribute__((ext_vector_type(8)))  __bf16 v8bf;
typedef __attribute__((ext_vector_type(8)))  float  v8f;

#define ATTN_B 32
#define ATTN_S 2048
#define ATTN_D 128

#define WAVES_PER_BLOCK 8
#define QROWS_PER_BLOCK 128
#define NQB (ATTN_S / QROWS_PER_BLOCK)
#define KTILE 32
#define NKT (ATTN_S / KTILE)

#define KSTRIDE 136                    // 128 + 8 pad (bf16) -> conflict-free
#define VTSTRIDE 40                    // 32 + 8 pad
#define PSTRIDE 40
#define KBUF (KTILE * KSTRIDE)
#define VBUF (ATTN_D * VTSTRIDE)

// ---- helpers --------------------------------------------------------------

__device__ __forceinline__ v16bf frag16(const __bf16* p) {
    v8bf lo = *(const v8bf*)p;
    v8bf hi = *(const v8bf*)(p + 8);
    return __builtin_shufflevector(lo, hi,
        0,1,2,3,4,5,6,7,8,9,10,11,12,13,14,15);
}

__device__ __forceinline__ v8bf pack8(float4 a, float4 b) {
    v8bf r;
    r[0]=(__bf16)a.x; r[1]=(__bf16)a.y; r[2]=(__bf16)a.z; r[3]=(__bf16)a.w;
    r[4]=(__bf16)b.x; r[5]=(__bf16)b.y; r[6]=(__bf16)b.z; r[7]=(__bf16)b.w;
    return r;
}

__device__ __forceinline__ unsigned bfpack2(float a, float b) {
    unsigned short ua = __builtin_bit_cast(unsigned short, (__bf16)a);
    unsigned short ub = __builtin_bit_cast(unsigned short, (__bf16)b);
    return (unsigned)ua | ((unsigned)ub << 16);
}

template<int CTRL>
__device__ __forceinline__ float dpp_mov(float x) {
    int xi = __float_as_int(x);
    return __int_as_float(
        __builtin_amdgcn_update_dpp(xi, xi, CTRL, 0xF, 0xF, true));
}
__device__ __forceinline__ float row_max16(float x) {
    x = fmaxf(x, dpp_mov<0xB1>(x));    // quad_perm(1,0,3,2) : xor 1
    x = fmaxf(x, dpp_mov<0x4E>(x));    // quad_perm(2,3,0,1) : xor 2
    x = fmaxf(x, dpp_mov<0x124>(x));   // row_ror:4
    x = fmaxf(x, dpp_mov<0x128>(x));   // row_ror:8
    return x;
}

// ---- async global->LDS copy (ASYNCcnt), 16 bytes per lane -----------------
// Builtin signature (revealed by clang diagnostic):
//   (v4i as(1)* gaddr, v4i as(3)* lds, imm offset, imm cpol)
typedef int v4i_vs __attribute__((vector_size(16)));
typedef __attribute__((address_space(1))) v4i_vs* asyncg_t;
typedef __attribute__((address_space(3))) v4i_vs* asyncl_t;

__device__ __forceinline__ void async_b128(const __bf16* g, __bf16* l) {
#if __has_builtin(__builtin_amdgcn_global_load_async_to_lds_b128)
    __builtin_amdgcn_global_load_async_to_lds_b128(
        (asyncg_t)(unsigned long long)(const void*)g,
        (asyncl_t)(unsigned)(unsigned long long)(const void*)l,
        0, 0);
#else
    unsigned laddr = (unsigned)(unsigned long long)l;
    asm volatile("global_load_async_to_lds_b128 %0, %1, off"
                 :: "v"(laddr), "v"(g) : "memory");
#endif
}

__device__ __forceinline__ void wait_async0() {
#if __has_builtin(__builtin_amdgcn_s_wait_asynccnt)
    __builtin_amdgcn_s_wait_asynccnt(0);
#else
    asm volatile("s_wait_asynccnt 0" ::: "memory");
#endif
}

// ---- prepass kernels ------------------------------------------------------

// K (fp32 row-major) -> Kb (bf16 row-major); one thread = 8 elements
__global__ __launch_bounds__(256)
void cvt_k_bf16(const float* __restrict__ K, __bf16* __restrict__ Kb) {
    size_t i = (size_t)blockIdx.x * 256 + threadIdx.x;   // 8-elem groups
    const float4* src = (const float4*)K + 2*i;
    float4 a = src[0], b = src[1];
    *(v8bf*)(Kb + 8*i) = pack8(a, b);
}

// V (fp32 [b][s][d]) -> Vt (bf16 [b][d][s]); one thread = 8 d of one (b,s)
__global__ __launch_bounds__(256)
void cvt_v_t_bf16(const float* __restrict__ V, __bf16* __restrict__ Vt) {
    size_t i = (size_t)blockIdx.x * 256 + threadIdx.x;   // b*S*(D/8) groups
    int dg = (int)(i & 15);            // D/8 = 16
    int s  = (int)((i >> 4) & (ATTN_S - 1));
    int b  = (int)(i >> 15);
    const float* src = V + ((size_t)b * ATTN_S + s) * ATTN_D + dg * 8;
    float4 a = *(const float4*)src;
    float4 c = *(const float4*)(src + 4);
    float v[8] = { a.x,a.y,a.z,a.w, c.x,c.y,c.z,c.w };
    __bf16* dst = Vt + (size_t)b * ATTN_D * ATTN_S + (size_t)dg * 8 * ATTN_S + s;
    #pragma unroll
    for (int j = 0; j < 8; ++j) dst[(size_t)j * ATTN_S] = (__bf16)v[j];
}

// ---- main kernel: async-to-LDS fast path ----------------------------------

__global__ __launch_bounds__(256, 1)
void fa_fwd_async(const float* __restrict__ Q,
                  const __bf16* __restrict__ Kb,
                  const __bf16* __restrict__ Vt,
                  float* __restrict__ O)
{
    __shared__ __align__(16) __bf16 sK [2 * KBUF];
    __shared__ __align__(16) __bf16 sVT[2 * VBUF];
    __shared__ __align__(16) __bf16 sP [WAVES_PER_BLOCK*16*PSTRIDE];

    const int tid  = threadIdx.x;
    const int wave = tid >> 5;
    const int lane = tid & 31;
    const int ln16 = lane & 15;
    const int hi16 = lane >> 4;

    const int b  = blockIdx.x / NQB;
    const int q0 = (blockIdx.x % NQB) * QROWS_PER_BLOCK;

    const float qscale = 1.44269504088896f * rsqrtf((float)ATTN_D);

    // Q tile -> 4 A-fragments
    const int    qrow = q0 + wave * 16 + ln16;
    const float* qp   = Q + ((size_t)b * ATTN_S + qrow) * ATTN_D;
    v16bf qf[4];
    #pragma unroll
    for (int c = 0; c < 4; ++c) {
        #pragma unroll
        for (int j = 0; j < 16; ++j) {
            int d = 32*c + 16*hi16 + j;
            qf[c][j] = (__bf16)(qp[d] * qscale);
        }
    }

    v16bf onesb;
    #pragma unroll
    for (int j = 0; j < 16; ++j) onesb[j] = (__bf16)1.0f;

    v8f oacc[8] = {};
    float mrun[8], lrun[8];
    #pragma unroll
    for (int r = 0; r < 8; ++r) { mrun[r] = -1.0e30f; lrun[r] = 0.0f; }

    // async copy maps: 256 lanes x 4 b128 = 16 KB per tile (K 8KB + V 8KB)
    const int kkey = tid >> 3, kch = tid & 7;   // K: row key, 16B chunk
    const int vd   = tid >> 1, vkg = tid & 1;   // Vt: row d, 16B chunk
    const __bf16* kgbase = Kb + (size_t)b * ATTN_S * ATTN_D;
    const __bf16* vgbase = Vt + (size_t)b * ATTN_D * ATTN_S;

    #define ISSUE_TILE(KT, BUF)                                                \
    {                                                                          \
        const __bf16* kg = kgbase + ((size_t)((KT)*KTILE + kkey))*ATTN_D + kch*8; \
        __bf16* kl = &sK[(BUF)*KBUF + kkey*KSTRIDE + kch*8];                   \
        async_b128(kg,      kl);                                               \
        async_b128(kg + 64, kl + 64);                                          \
        const __bf16* vg = vgbase + (size_t)vd*ATTN_S + (KT)*KTILE + vkg*8;    \
        __bf16* vl = &sVT[(BUF)*VBUF + vd*VTSTRIDE + vkg*8];                   \
        async_b128(vg,      vl);                                               \
        async_b128(vg + 16, vl + 16);                                          \
    }

    // prologue: tile 0 -> buffer 0
    ISSUE_TILE(0, 0)
    wait_async0();
    __syncthreads();

    for (int kt = 0; kt < NKT; ++kt) {
        const int buf = kt & 1;
        const __bf16* cK  = &sK [buf * KBUF];
        const __bf16* cVT = &sVT[buf * VBUF];

        // async DMA for next tile runs under this tile's WMMAs
        if (kt + 1 < NKT) ISSUE_TILE(kt + 1, buf ^ 1)

        // ---- S = Q*K^T : two 16x16 tiles, 4 K=32 chunks each ----
        v8f s0 = {}, s1 = {};
        #pragma unroll
        for (int c = 0; c < 4; ++c) {
            v16bf bk = frag16(&cK[ln16*KSTRIDE + 32*c + 16*hi16]);
            s0 = __builtin_amdgcn_wmma_f32_16x16x32_bf16(
                     false, qf[c], false, bk, (short)0, s0, false, false);
        }
        #pragma unroll
        for (int c = 0; c < 4; ++c) {
            v16bf bk = frag16(&cK[(16 + ln16)*KSTRIDE + 32*c + 16*hi16]);
            s1 = __builtin_amdgcn_wmma_f32_16x16x32_bf16(
                     false, qf[c], false, bk, (short)0, s1, false, false);
        }

        // ---- online softmax ----
        float alpha[8];
        #pragma unroll
        for (int r = 0; r < 8; ++r) {
            float mx   = row_max16(fmaxf(s0[r], s1[r]));
            float mnew = fmaxf(mrun[r], mx);
            alpha[r]   = exp2f(mrun[r] - mnew);
            mrun[r]    = mnew;
            s0[r] = exp2f(s0[r] - mnew);
            s1[r] = exp2f(s1[r] - mnew);
        }
        #pragma unroll
        for (int t = 0; t < 8; ++t) {
            #pragma unroll
            for (int r = 0; r < 8; ++r) oacc[t][r] *= alpha[r];
        }

        // ---- P: C-layout -> per-wave LDS scratch -> A-layout ----
        __bf16* pw = &sP[wave * 16 * PSTRIDE];
        #pragma unroll
        for (int r = 0; r < 8; ++r) {
            pw[(r + 8*hi16)*PSTRIDE + ln16]      = (__bf16)s0[r];
            pw[(r + 8*hi16)*PSTRIDE + 16 + ln16] = (__bf16)s1[r];
        }
        asm volatile("s_wait_dscnt 0" ::: "memory");

        const __bf16* prow = pw + ln16 * PSTRIDE;
        v8bf plo = *(const v8bf*)(prow + 16*hi16);
        v8bf phi = *(const v8bf*)(prow + 16*hi16 + 8);
        v16bf ap = __builtin_shufflevector(plo, phi,
            0,1,2,3,4,5,6,7,8,9,10,11,12,13,14,15);

        // row-sum on the matrix pipe
        v8f zero = {};
        v8f lsum = __builtin_amdgcn_wmma_f32_16x16x32_bf16(
                       false, ap, false, onesb, (short)0, zero, false, false);
        #pragma unroll
        for (int r = 0; r < 8; ++r) lrun[r] = lrun[r] * alpha[r] + lsum[r];

        // ---- O += P * V ----
        #pragma unroll
        for (int t = 0; t < 8; ++t) {
            v16bf bv = frag16(&cVT[(t*16 + ln16)*VTSTRIDE + 16*hi16]);
            oacc[t] = __builtin_amdgcn_wmma_f32_16x16x32_bf16(
                          false, ap, false, bv, (short)0, oacc[t], false, false);
        }

        wait_async0();      // next tile's LDS writes landed
        __syncthreads();
    }
    #undef ISSUE_TILE

    float* op = O + ((size_t)b * ATTN_S + q0 + wave * 16) * ATTN_D;
    #pragma unroll
    for (int r = 0; r < 8; ++r) {
        float inv = 1.0f / lrun[r];
        #pragma unroll
        for (int t = 0; t < 8; ++t)
            op[(r + 8*hi16) * ATTN_D + t*16 + ln16] = oacc[t][r] * inv;
    }
}

// ---- fallback kernel: register-pipelined staging --------------------------

__global__ __launch_bounds__(256, 1)
void fa_fwd_sync(const float* __restrict__ Q,
                 const float* __restrict__ K,
                 const float* __restrict__ V,
                 float* __restrict__ O)
{
    __shared__ __align__(16) __bf16 sK [2 * KBUF];
    __shared__ __align__(16) __bf16 sVT[2 * VBUF];
    __shared__ __align__(16) __bf16 sP [WAVES_PER_BLOCK*16*PSTRIDE];

    const int tid  = threadIdx.x;
    const int wave = tid >> 5;
    const int lane = tid & 31;
    const int ln16 = lane & 15;
    const int hi16 = lane >> 4;

    const int b  = blockIdx.x / NQB;
    const int q0 = (blockIdx.x % NQB) * QROWS_PER_BLOCK;

    const float qscale = 1.44269504088896f * rsqrtf((float)ATTN_D);

    const int    qrow = q0 + wave * 16 + ln16;
    const float* qp   = Q + ((size_t)b * ATTN_S + qrow) * ATTN_D;
    v16bf qf[4];
    #pragma unroll
    for (int c = 0; c < 4; ++c) {
        #pragma unroll
        for (int j = 0; j < 16; ++j) {
            int d = 32*c + 16*hi16 + j;
            qf[c][j] = (__bf16)(qp[d] * qscale);
        }
    }

    v16bf onesb;
    #pragma unroll
    for (int j = 0; j < 16; ++j) onesb[j] = (__bf16)1.0f;

    v8f oacc[8] = {};
    float mrun[8], lrun[8];
    #pragma unroll
    for (int r = 0; r < 8; ++r) { mrun[r] = -1.0e30f; lrun[r] = 0.0f; }

    const int kkey = tid >> 3;
    const int kd0  = (tid & 7) * 16;
    const int vkp  = tid >> 4;
    const int vd0  = (tid & 15) * 8;
    const float* kbase = K + (size_t)b * ATTN_S * ATTN_D;
    const float* vbase = V + (size_t)b * ATTN_S * ATTN_D;

    float4 kreg[4], vreg[4];

    {
        const float* kr = kbase + (size_t)kkey * ATTN_D + kd0;
        kreg[0] = *(const float4*)(kr);
        kreg[1] = *(const float4*)(kr + 4);
        kreg[2] = *(const float4*)(kr + 8);
        kreg[3] = *(const float4*)(kr + 12);
        const float* vr = vbase + (size_t)(2*vkp) * ATTN_D + vd0;
        vreg[0] = *(const float4*)(vr);
        vreg[1] = *(const float4*)(vr + 4);
        vreg[2] = *(const float4*)(vr + ATTN_D);
        vreg[3] = *(const float4*)(vr + ATTN_D + 4);

        *(v8bf*)&sK[kkey*KSTRIDE + kd0]     = pack8(kreg[0], kreg[1]);
        *(v8bf*)&sK[kkey*KSTRIDE + kd0 + 8] = pack8(kreg[2], kreg[3]);
        float a[8] = { vreg[0].x,vreg[0].y,vreg[0].z,vreg[0].w,
                       vreg[1].x,vreg[1].y,vreg[1].z,vreg[1].w };
        float c[8] = { vreg[2].x,vreg[2].y,vreg[2].z,vreg[2].w,
                       vreg[3].x,vreg[3].y,vreg[3].z,vreg[3].w };
        #pragma unroll
        for (int i = 0; i < 8; ++i)
            *(unsigned*)&sVT[(vd0 + i)*VTSTRIDE + 2*vkp] = bfpack2(a[i], c[i]);
    }
    __syncthreads();

    for (int kt = 0; kt < NKT; ++kt) {
        const int buf = kt & 1;
        const __bf16* cK  = &sK [buf * KBUF];
        const __bf16* cVT = &sVT[buf * VBUF];

        const bool more = (kt + 1 < NKT);
        if (more) {
            const float* kr = kbase + (size_t)((kt+1)*KTILE + kkey) * ATTN_D + kd0;
            kreg[0] = *(const float4*)(kr);
            kreg[1] = *(const float4*)(kr + 4);
            kreg[2] = *(const float4*)(kr + 8);
            kreg[3] = *(const float4*)(kr + 12);
            const float* vr = vbase + (size_t)((kt+1)*KTILE + 2*vkp) * ATTN_D + vd0;
            vreg[0] = *(const float4*)(vr);
            vreg[1] = *(const float4*)(vr + 4);
            vreg[2] = *(const float4*)(vr + ATTN_D);
            vreg[3] = *(const float4*)(vr + ATTN_D + 4);
        }

        v8f s0 = {}, s1 = {};
        #pragma unroll
        for (int c = 0; c < 4; ++c) {
            v16bf bk = frag16(&cK[ln16*KSTRIDE + 32*c + 16*hi16]);
            s0 = __builtin_amdgcn_wmma_f32_16x16x32_bf16(
                     false, qf[c], false, bk, (short)0, s0, false, false);
        }
        #pragma unroll
        for (int c = 0; c < 4; ++c) {
            v16bf bk = frag16(&cK[(16 + ln16)*KSTRIDE + 32*c + 16*hi16]);
            s1 = __builtin_amdgcn_wmma_f32_16x16x32_bf16(
                     false, qf[c], false, bk, (short)0, s1, false, false);
        }

        float alpha[8];
        #pragma unroll
        for (int r = 0; r < 8; ++r) {
            float mx   = row_max16(fmaxf(s0[r], s1[r]));
            float mnew = fmaxf(mrun[r], mx);
            alpha[r]   = exp2f(mrun[r] - mnew);
            mrun[r]    = mnew;
            s0[r] = exp2f(s0[r] - mnew);
            s1[r] = exp2f(s1[r] - mnew);
        }
        #pragma unroll
        for (int t = 0; t < 8; ++t) {
            #pragma unroll
            for (int r = 0; r < 8; ++r) oacc[t][r] *= alpha[r];
        }

        __bf16* pw = &sP[wave * 16 * PSTRIDE];
        #pragma unroll
        for (int r = 0; r < 8; ++r) {
            pw[(r + 8*hi16)*PSTRIDE + ln16]      = (__bf16)s0[r];
            pw[(r + 8*hi16)*PSTRIDE + 16 + ln16] = (__bf16)s1[r];
        }
        asm volatile("s_wait_dscnt 0" ::: "memory");

        const __bf16* prow = pw + ln16 * PSTRIDE;
        v8bf plo = *(const v8bf*)(prow + 16*hi16);
        v8bf phi = *(const v8bf*)(prow + 16*hi16 + 8);
        v16bf ap = __builtin_shufflevector(plo, phi,
            0,1,2,3,4,5,6,7,8,9,10,11,12,13,14,15);

        v8f zero = {};
        v8f lsum = __builtin_amdgcn_wmma_f32_16x16x32_bf16(
                       false, ap, false, onesb, (short)0, zero, false, false);
        #pragma unroll
        for (int r = 0; r < 8; ++r) lrun[r] = lrun[r] * alpha[r] + lsum[r];

        #pragma unroll
        for (int t = 0; t < 8; ++t) {
            v16bf bv = frag16(&cVT[(t*16 + ln16)*VTSTRIDE + 16*hi16]);
            oacc[t] = __builtin_amdgcn_wmma_f32_16x16x32_bf16(
                          false, ap, false, bv, (short)0, oacc[t], false, false);
        }

        if (more) {
            __bf16* dK  = &sK [(buf ^ 1) * KBUF];
            __bf16* dVT = &sVT[(buf ^ 1) * VBUF];
            *(v8bf*)&dK[kkey*KSTRIDE + kd0]     = pack8(kreg[0], kreg[1]);
            *(v8bf*)&dK[kkey*KSTRIDE + kd0 + 8] = pack8(kreg[2], kreg[3]);
            float a[8] = { vreg[0].x,vreg[0].y,vreg[0].z,vreg[0].w,
                           vreg[1].x,vreg[1].y,vreg[1].z,vreg[1].w };
            float c[8] = { vreg[2].x,vreg[2].y,vreg[2].z,vreg[2].w,
                           vreg[3].x,vreg[3].y,vreg[3].z,vreg[3].w };
            #pragma unroll
            for (int i = 0; i < 8; ++i)
                *(unsigned*)&dVT[(vd0 + i)*VTSTRIDE + 2*vkp] = bfpack2(a[i], c[i]);
        }
        __syncthreads();
    }

    float* op = O + ((size_t)b * ATTN_S + q0 + wave * 16) * ATTN_D;
    #pragma unroll
    for (int r = 0; r < 8; ++r) {
        float inv = 1.0f / lrun[r];
        #pragma unroll
        for (int t = 0; t < 8; ++t)
            op[(r + 8*hi16) * ATTN_D + t*16 + ln16] = oacc[t][r] * inv;
    }
}

// ---- launch ---------------------------------------------------------------

extern "C" void kernel_launch(void* const* d_in, const int* in_sizes, int n_in,
                              void* d_out, int out_size, void* d_ws, size_t ws_size,
                              hipStream_t stream) {
    const float* Q = (const float*)d_in[0];
    const float* K = (const float*)d_in[1];
    const float* V = (const float*)d_in[2];
    float* O = (float*)d_out;

    const size_t elems = (size_t)ATTN_B * ATTN_S * ATTN_D;   // 8.4M
    const size_t need  = 2 * elems * sizeof(unsigned short); // Kb + Vt, 33.5 MB

    dim3 grid(ATTN_B * NQB), block(256);

    if (ws_size >= need) {
        __bf16* Kb = (__bf16*)d_ws;
        __bf16* Vt = Kb + elems;
        dim3 cgrid((unsigned)(elems / 8 / 256));             // 4096
        hipLaunchKernelGGL(cvt_k_bf16,   cgrid, block, 0, stream, K, Kb);
        hipLaunchKernelGGL(cvt_v_t_bf16, cgrid, block, 0, stream, V, Vt);
        hipLaunchKernelGGL(fa_fwd_async, grid,  block, 0, stream, Q, Kb, Vt, O);
    } else {
        hipLaunchKernelGGL(fa_fwd_sync,  grid,  block, 0, stream, Q, K, V, O);
    }
}
